// Edge_edge_50869592655542
// MI455X (gfx1250) — matrix operand order repaired and checked by reading, not compiled, not run on hardware
//
#include <hip/hip_runtime.h>

// ---------- types ----------
typedef __attribute__((ext_vector_type(16))) __bf16 v16bf;
typedef __attribute__((ext_vector_type(8)))  float  v8f;
typedef __attribute__((ext_vector_type(4)))  unsigned v4u;
typedef __attribute__((ext_vector_type(8)))  int   v8i;
typedef __attribute__((ext_vector_type(4)))  int   v4i;

union Frag32B { uint4 q[2]; v16bf v; };

__device__ __forceinline__ unsigned short f32_to_bf16(float f) {
  union { float f; unsigned u; } v; v.f = f;
  unsigned u = v.u;
  unsigned r = u + 0x7FFFu + ((u >> 16) & 1u);   // round-to-nearest-even
  return (unsigned short)(r >> 16);
}

// ---------- utility kernels ----------
__global__ void zero_n(float* __restrict__ p, long long n) {
  long long i = (long long)blockIdx.x * blockDim.x + threadIdx.x;
  if (i < n) p[i] = 0.f;
}

// w [K,N] f32 row-major -> wt [N,K] bf16 row-major
__global__ void wt_to_bf16(const float* __restrict__ w, unsigned short* __restrict__ wt,
                           int K, int N) {
  int i = blockIdx.x * blockDim.x + threadIdx.x;
  if (i >= K * N) return;
  int k = i / N, n = i % N;
  wt[(long long)n * K + k] = f32_to_bf16(w[i]);
}

// edge_rep [M,C] f32 -> feat[:,0:C] bf16 (feat leading dim = 5C)
__global__ void copy_feat_bf16(const float* __restrict__ x, unsigned short* __restrict__ feat,
                               int c, int ld) {
  long long r = blockIdx.x; int j = threadIdx.x;
  feat[r * ld + j] = f32_to_bf16(x[r * c + j]);
}

// segment-sum: S[node[r], j] += x[r, j]
__global__ void scatter_add(const float* __restrict__ x, const int* __restrict__ node,
                            float* __restrict__ S, int c) {
  long long r = blockIdx.x; int j = threadIdx.x;
  atomicAdd(&S[(long long)node[r] * c + j], x[r * (long long)c + j]);
}

// m1[r] = [ S[node[r]] , S[node[r]] + x[r^1] ]   (chB = Sa - x + pair_sum simplification)
__global__ void gather_round1(const float* __restrict__ x, const int* __restrict__ node,
                              const float* __restrict__ S, float* __restrict__ m1, int c) {
  long long r = blockIdx.x; int j = threadIdx.x;
  float sa = S[(long long)node[r] * c + j];
  float xp = x[(r ^ 1) * (long long)c + j];
  float* o = m1 + r * (long long)(2 * c);
  o[j]     = sa;
  o[c + j] = sa + xp;
}

// second gather writes directly into feat columns [C, 5C) as bf16
__global__ void gather_round2_feat(const float* __restrict__ m1, const int* __restrict__ node,
                                   const float* __restrict__ S2, unsigned short* __restrict__ feat,
                                   int c2, int ld, int coff) {
  long long r = blockIdx.x; int j = threadIdx.x;     // blockDim = c2 = 2C
  float sa = S2[(long long)node[r] * c2 + j];
  float xp = m1[(r ^ 1) * (long long)c2 + j];
  unsigned short* o = feat + r * (long long)ld;
  o[coff + j]       = f32_to_bf16(sa);
  o[coff + c2 + j]  = f32_to_bf16(sa + xp);
}

// ---------- TDM staging of a 128 x K bf16 panel into LDS (padded rows) ----------
// padded LDS row stride (bytes) for a K-element bf16 row with 16B pad per 256B:
__device__ __host__ __forceinline__ unsigned b_row_stride(int K) {
  return 2u * (unsigned)K + ((unsigned)K / 128u) * 16u;
}

__device__ __forceinline__ void tdm_stage_b(const unsigned short* gsrc, void* lds, int K) {
#if __has_builtin(__builtin_amdgcn_tensor_load_to_lds)
  unsigned long long ga = (unsigned long long)gsrc;
  v4u g0;
  g0.x = 1u;                                            // count=1, user mode
  g0.y = (unsigned)(unsigned long long)lds;             // lds_addr (low 32 bits = LDS offset)
  g0.z = (unsigned)(ga & 0xFFFFFFFFull);                // global_addr[31:0]
  g0.w = (unsigned)((ga >> 32) & 0x01FFFFFFull) | (2u << 30);  // addr[56:32] | type=2
  v8i g1;
  // data_size=1 (2B) | pad_enable | pad_interval=5 (64 DW = 256B) | pad_amount=3 (4 DW = 16B)
  g1[0] = (int)((1u << 16) | (1u << 20) | (5u << 22) | (3u << 25));
  g1[1] = (int)(((unsigned)K & 0xFFFFu) << 16);         // tensor_dim0[15:0]
  g1[2] = (int)((((unsigned)K >> 16) & 0xFFFFu) | (128u << 16)); // dim0[31:16] | tensor_dim1[15:0]=128
  g1[3] = (int)(((unsigned)K & 0xFFFFu) << 16);         // tensor_dim1[31:16]=0 | tile_dim0=K
  g1[4] = (int)128;                                     // tile_dim1=128, tile_dim2=0
  g1[5] = (int)K;                                       // tensor_dim0_stride[31:0]=K
  g1[6] = 0;                                            // stride hi | dim1_stride lo
  g1[7] = 0;
  v4i z4 = {0, 0, 0, 0};
#if __clang_major__ >= 23
  v8i z8 = {0, 0, 0, 0, 0, 0, 0, 0};
  __builtin_amdgcn_tensor_load_to_lds(g0, g1, z4, z4, z8, 0);
#else
  __builtin_amdgcn_tensor_load_to_lds(g0, g1, z4, z4, 0);
#endif
  __builtin_amdgcn_s_wait_tensorcnt(0);
#else
  (void)gsrc; (void)lds; (void)K;
#endif
}

// ---------- WMMA GEMM: Y[M,N] f32 = A[M,K] bf16 @ Bt[N,K]^T bf16 ----------
// Block = 4 waves on 4 M-tiles; the block's 128-column B panel is TDM-staged to LDS
// (padded rows, conflict-spread) and every WMMA reads B via ds_load_b128.
template<int NT>
__global__ __launch_bounds__(128) void wmma_gemm_bf16(
    const unsigned short* __restrict__ A, const unsigned short* __restrict__ Bt,
    float* __restrict__ Y, long long M, int N, int K) {
  extern __shared__ char smem[];
  const int wave = threadIdx.x >> 5;
  const int lane = threadIdx.x & 31;
  const int half = lane >> 4;
  const int l16  = lane & 15;
  const long long m0 = ((long long)blockIdx.x * 4 + wave) * 16;
  const long long n0 = (long long)blockIdx.y * (16 * NT);
  const unsigned rowStride = b_row_stride(K);

#if __has_builtin(__builtin_amdgcn_tensor_load_to_lds)
  if (threadIdx.x < 32)                       // one TDM issue per block (EXEC-ignored op)
    tdm_stage_b(Bt + n0 * (long long)K, smem, K);
#else
  for (unsigned idx = threadIdx.x; idx < (unsigned)(128 * K / 8); idx += blockDim.x) {
    unsigned n = idx / ((unsigned)K / 8u);
    unsigned ke = (idx % ((unsigned)K / 8u)) * 8u;
    *(uint4*)(smem + n * rowStride + ke * 2u + (ke >> 7) * 16u) =
        *(const uint4*)(Bt + (n0 + n) * (long long)K + ke);
  }
#endif
  __syncthreads();
  if (m0 + 16 > M) return;                    // wave-uniform; EXEC stays all-1s

  const unsigned short* arow = A + (m0 + l16) * (long long)K;

  v8f acc[NT];
#pragma unroll
  for (int t = 0; t < NT; ++t)
#pragma unroll
    for (int r = 0; r < 8; ++r) acc[t][r] = 0.f;

  for (int k = 0; k < K; k += 32) {
    // A fragment (this code's contiguous k order, same as B): lanes<16 hold K=[k..k+15]; lanes>=16 hold K=[k+16..k+31]
    const unsigned short* ap = arow + k + half * 16;
    __builtin_prefetch(ap + 64, 0, 1);
    Frag32B a;
    a.q[0] = *(const uint4*)(ap);
    a.q[1] = *(const uint4*)(ap + 8);
#pragma unroll
    for (int t = 0; t < NT; ++t) {
      // B fragment: lane = column, lanes<16 K=k..k+15, lanes>=16 K=k+16..k+31 (from LDS)
      unsigned ke  = (unsigned)(k + half * 16);
      unsigned off = (unsigned)(t * 16 + l16) * rowStride + ke * 2u + (ke >> 7) * 16u;
      const uint4* bp = (const uint4*)(smem + off);
      Frag32B b;
      b.q[0] = bp[0];
      b.q[1] = bp[1];
      acc[t] = __builtin_amdgcn_wmma_f32_16x16x32_bf16(
          false, a.v, false, b.v, (short)0, acc[t], false, false);
    }
  }

#pragma unroll
  for (int t = 0; t < NT; ++t) {
    long long col = n0 + t * 16 + l16;
#pragma unroll
    for (int r = 0; r < 8; ++r) {
      long long row = m0 + r + half * 8;         // C/D f32 16x16 layout
      Y[row * N + col] = acc[t][r];
    }
  }
}

// ---------- BatchNorm (training-mode) ----------
__global__ void bn_stats(const float* __restrict__ y, float* __restrict__ stats,
                         long long M, int NC) {
  int j = threadIdx.x;                            // blockDim = NC
  float s = 0.f, sq = 0.f;
  for (long long r = blockIdx.x; r < M; r += gridDim.x) {
    float v = y[r * NC + j];
    s += v; sq += v * v;
  }
  atomicAdd(&stats[j], s);
  atomicAdd(&stats[NC + j], sq);
}

__global__ void bn_relu_to_bf16(const float* __restrict__ y, const float* __restrict__ stats,
                                const float* __restrict__ g, const float* __restrict__ b,
                                unsigned short* __restrict__ h, long long M, int NC) {
  long long r = blockIdx.x; int j = threadIdx.x;
  float mean = stats[j] / (float)M;
  float var  = stats[NC + j] / (float)M - mean * mean;
  float rs   = rsqrtf(var + 1e-5f);
  float v = (y[r * NC + j] - mean) * rs * g[j] + b[j];
  h[r * NC + j] = f32_to_bf16(fmaxf(v, 0.f));
}

__global__ void bn_relu_inplace(float* __restrict__ y, const float* __restrict__ stats,
                                const float* __restrict__ g, const float* __restrict__ b,
                                long long M, int NC) {
  long long r = blockIdx.x; int j = threadIdx.x;
  float mean = stats[j] / (float)M;
  float var  = stats[NC + j] / (float)M - mean * mean;
  float rs   = rsqrtf(var + 1e-5f);
  float v = (y[r * NC + j] - mean) * rs * g[j] + b[j];
  y[r * NC + j] = fmaxf(v, 0.f);
}

// ---------- launch ----------
extern "C" void kernel_launch(void* const* d_in, const int* in_sizes, int n_in,
                              void* d_out, int out_size, void* d_ws, size_t ws_size,
                              hipStream_t stream) {
  const float* edge_rep = (const float*)d_in[0];
  const float* w1 = (const float*)d_in[1];
  const float* g1 = (const float*)d_in[2];
  const float* b1 = (const float*)d_in[3];
  const float* w2 = (const float*)d_in[4];
  const float* g2 = (const float*)d_in[5];
  const float* b2 = (const float*)d_in[6];
  const int* node = (const int*)d_in[7];          // edge_index flattened = node ids per row
  (void)n_in; (void)ws_size;

  const int C  = in_sizes[5];                     // g2 has C elements
  const long long M = (long long)in_sizes[0] / C; // 2E rows
  const int C2 = 2 * C, C5 = 5 * C;
  const long long NB = M;                         // bound on n_nodes (N=50000 << 2E here)

  char* p = (char*)d_ws;
  auto carve = [&](size_t bytes) -> char* {
    char* q = p; p += (bytes + 255) & ~(size_t)255; return q;
  };
  float*          S1   = (float*)carve((size_t)NB * C  * 4);
  float*          S2   = (float*)carve((size_t)NB * C2 * 4);
  float*          m1   = (float*)carve((size_t)M  * C2 * 4);   // reused as y1
  unsigned short* feat = (unsigned short*)carve((size_t)M * C5 * 2);
  unsigned short* h    = (unsigned short*)carve((size_t)M * C2 * 2);
  unsigned short* w1t  = (unsigned short*)carve((size_t)C5 * C2 * 2);
  unsigned short* w2t  = (unsigned short*)carve((size_t)C2 * C  * 2);
  float*          st1  = (float*)carve((size_t)2 * C2 * 4);
  float*          st2  = (float*)carve((size_t)2 * C  * 4);
  float* y1  = m1;
  float* out = (float*)d_out;

  const int TB = 256;
  auto nb = [&](long long n) { return (unsigned)((n + TB - 1) / TB); };

  zero_n<<<nb(NB * C),  TB, 0, stream>>>(S1, NB * C);
  zero_n<<<nb(NB * C2), TB, 0, stream>>>(S2, NB * C2);
  zero_n<<<nb(2 * C2),  TB, 0, stream>>>(st1, 2 * C2);
  zero_n<<<nb(2 * C),   TB, 0, stream>>>(st2, 2 * C);

  wt_to_bf16<<<nb((long long)C5 * C2), TB, 0, stream>>>(w1, w1t, C5, C2);
  wt_to_bf16<<<nb((long long)C2 * C),  TB, 0, stream>>>(w2, w2t, C2, C);

  copy_feat_bf16<<<(unsigned)M, C, 0, stream>>>(edge_rep, feat, C, C5);

  // round 1
  scatter_add<<<(unsigned)M, C, 0, stream>>>(edge_rep, node, S1, C);
  gather_round1<<<(unsigned)M, C, 0, stream>>>(edge_rep, node, S1, m1, C);
  // round 2 (emit bf16 straight into feat[:, C:5C))
  scatter_add<<<(unsigned)M, C2, 0, stream>>>(m1, node, S2, C2);
  gather_round2_feat<<<(unsigned)M, C2, 0, stream>>>(m1, node, S2, feat, C2, C5, C);

  // GEMM1: y1[M,2C] = feat[M,5C] @ w1  (y1 overwrites m1 — m1 is dead now)
  {
    dim3 grid((unsigned)(M / 64), (unsigned)(C2 / 128));
    unsigned smemB = 128u * b_row_stride(C5);      // padded B panel
    wmma_gemm_bf16<8><<<grid, 128, smemB, stream>>>(feat, w1t, y1, M, C2, C5);
  }
  bn_stats<<<1024, C2, 0, stream>>>(y1, st1, M, C2);
  bn_relu_to_bf16<<<(unsigned)M, C2, 0, stream>>>(y1, st1, g1, b1, h, M, C2);

  // GEMM2: out[M,C] = h[M,2C] @ w2
  {
    dim3 grid((unsigned)(M / 64), (unsigned)(C / 128));
    unsigned smemB = 128u * b_row_stride(C2);
    wmma_gemm_bf16<8><<<grid, 128, smemB, stream>>>(h, w2t, out, M, C, C2);
  }
  bn_stats<<<1024, C, 0, stream>>>(out, st2, M, C);
  bn_relu_inplace<<<(unsigned)M, C, 0, stream>>>(out, st2, g2, b2, M, C);
  (void)out_size;
}
